// _DinatLayerNd_49993419325476
// MI455X (gfx1250) — matrix operand order repaired and checked by reading, not compiled, not run on hardware
//
#include <hip/hip_runtime.h>
#include <hip/hip_bf16.h>

// ---------------------------------------------------------------------------
// Problem constants (from reference)
// ---------------------------------------------------------------------------
#define N_    4
#define T_    4096
#define C_    512
#define H_    8
#define DH_   64
#define KSZ_  7
#define DIL_  4
#define FF_   2048
#define NT_   (N_ * T_)
#define L_    (T_ / DIL_)

// GEMM tiling
#define BM_   128
#define BN_   128
#define BK_   32
#define LDT_  40      // padded LDS row stride in halves (32 data + 8 pad)

typedef __attribute__((ext_vector_type(16))) _Float16 v16h;
typedef __attribute__((ext_vector_type(8)))  _Float16 v8h;
typedef __attribute__((ext_vector_type(8)))  float    v8f;

// Workgroup-relative LDS byte offset: flat shared-aperture addresses keep the
// LDS offset in addr[31:0] (CDNA5 ISA 10.2 aperture mapping).
template <typename T>
__device__ __forceinline__ unsigned as_lds_off(T* p) {
    return (unsigned)(size_t)(p);
}

// ---------------------------------------------------------------------------
// Async global -> LDS tile copy (128 rows x 32 halves), 256 threads.
// 512 x 16B chunks; each thread issues two global_load_async_to_lds_b128.
// Tracked by ASYNCcnt (waited with s_wait_asynccnt).
// ---------------------------------------------------------------------------
__device__ __forceinline__ void async_tile_copy(const _Float16* __restrict__ g,
                                                int ld, _Float16* lsdst) {
    const int tid = threadIdx.x;
#pragma unroll
    for (int i = 0; i < 2; ++i) {
        const int c   = tid + i * 256;      // chunk id 0..511 (coalesced)
        const int row = c >> 2;             // 0..127
        const int qo  = (c & 3) << 3;       // half offset: 0,8,16,24
        const _Float16* gp = g + (size_t)row * ld + qo;
        const unsigned  lo = as_lds_off(lsdst + row * LDT_ + qo);
        asm volatile("global_load_async_to_lds_b128 %0, %1, off"
                     :: "v"(lo), "v"(gp)
                     : "memory");
    }
}

__device__ __forceinline__ void wait_async_zero() {
    asm volatile("s_wait_asynccnt 0" ::: "memory");
}

// ---------------------------------------------------------------------------
// WMMA fragment loaders from LDS tile (wave32 layouts per CDNA5 ISA 7.12.2)
// ---------------------------------------------------------------------------
// A 16x32 f16 (MxK): lanes 0-15 -> M=lane, halves K=0..7 then 16..23;
//                    lanes 16-31 -> M=lane-16, halves K=8..15 then 24..31.
__device__ __forceinline__ v16h lds_frag_a(const _Float16* tile) {
    const int lane = threadIdx.x & 31;
    const int row  = lane & 15;
    const int kb   = (lane >> 4) << 3;      // 0 or 8
    const _Float16* p = tile + row * LDT_ + kb;
    v8h lo = *(const v8h*)(p);
    v8h hi = *(const v8h*)(p + 16);
    v16h r;
#pragma unroll
    for (int i = 0; i < 8; ++i) { r[i] = lo[i]; r[i + 8] = hi[i]; }
    return r;
}

// B 32x16 f16 (KxN), source W[N,K]: lanes 0-15 -> N=lane, K=0..7 then 16..23;
//                                    lanes 16-31 -> N=lane-16, K=8..15 then 24..31.
__device__ __forceinline__ v16h lds_frag_b(const _Float16* tile) {
    const int lane = threadIdx.x & 31;
    const int col  = lane & 15;
    const int kb   = (lane >> 4) << 3;      // 0 or 8
    const _Float16* p = tile + col * LDT_ + kb;
    v8h lo = *(const v8h*)(p);
    v8h hi = *(const v8h*)(p + 16);
    v16h r;
#pragma unroll
    for (int i = 0; i < 8; ++i) { r[i] = lo[i]; r[i + 8] = hi[i]; }
    return r;
}

// ---------------------------------------------------------------------------
// WMMA GEMM with async-LDS double buffering:
//   out[M,Nd] = A[M,K](f16) * B[Nd,K](f16)^T + epilogue
// Block: 256 threads = 8 waves; wave tile 32x64 (2x4 accs); block tile 128x128.
// EPI 0: out_f16 = acc + bias                 (QKV)
// EPI 1: out_f32 = acc + bias + residual      (Wo / W2 + skip)
// EPI 2: out_f16 = gelu_exact(acc + bias)     (W1)
// ---------------------------------------------------------------------------
template <int EPI>
__global__ __launch_bounds__(256)
void gemm_wmma_kernel(const _Float16* __restrict__ A,
                      const _Float16* __restrict__ B,
                      const float*    __restrict__ bias,
                      const float*    __restrict__ residual,
                      float*          __restrict__ outF,
                      _Float16*       __restrict__ outH,
                      int M, int Nd, int K)
{
    __shared__ _Float16 lsA[2][BM_ * LDT_];
    __shared__ _Float16 lsB[2][BN_ * LDT_];

    const int wave = threadIdx.x >> 5;
    const int wm   = wave & 3;                  // 0..3 along M
    const int wn   = wave >> 2;                 // 0..1 along N
    const int bm0  = blockIdx.x * BM_;
    const int bn0  = blockIdx.y * BN_;
    const int m0   = bm0 + wm * 32;
    const int n0   = bn0 + wn * 64;

    v8f acc[2][4];
#pragma unroll
    for (int i = 0; i < 2; ++i)
#pragma unroll
        for (int j = 0; j < 4; ++j)
            acc[i][j] = (v8f){0.f, 0.f, 0.f, 0.f, 0.f, 0.f, 0.f, 0.f};

    const int nk = K / BK_;

    // Prologue: stage tile 0
    async_tile_copy(A + (size_t)bm0 * K, K, &lsA[0][0]);
    async_tile_copy(B + (size_t)bn0 * K, K, &lsB[0][0]);
    wait_async_zero();
    __syncthreads();

    for (int kt = 0; kt < nk; ++kt) {
        const int cur = kt & 1;

        // Prefetch next K-slab into the other buffer (overlaps with WMMA below)
        if (kt + 1 < nk) {
            const size_t ko = (size_t)(kt + 1) * BK_;
            async_tile_copy(A + (size_t)bm0 * K + ko, K, &lsA[1 - cur][0]);
            async_tile_copy(B + (size_t)bn0 * K + ko, K, &lsB[1 - cur][0]);
        }

        const _Float16* at = &lsA[cur][(wm * 32) * LDT_];
        const _Float16* bt = &lsB[cur][(wn * 64) * LDT_];

        v16h a0 = lds_frag_a(at);
        v16h a1 = lds_frag_a(at + 16 * LDT_);
        v16h b0 = lds_frag_b(bt);
        v16h b1 = lds_frag_b(bt + 16 * LDT_);
        v16h b2 = lds_frag_b(bt + 32 * LDT_);
        v16h b3 = lds_frag_b(bt + 48 * LDT_);

        acc[0][0] = __builtin_amdgcn_wmma_f32_16x16x32_f16(false, a0, false, b0, (short)0, acc[0][0], false, false);
        acc[0][1] = __builtin_amdgcn_wmma_f32_16x16x32_f16(false, a0, false, b1, (short)0, acc[0][1], false, false);
        acc[0][2] = __builtin_amdgcn_wmma_f32_16x16x32_f16(false, a0, false, b2, (short)0, acc[0][2], false, false);
        acc[0][3] = __builtin_amdgcn_wmma_f32_16x16x32_f16(false, a0, false, b3, (short)0, acc[0][3], false, false);
        acc[1][0] = __builtin_amdgcn_wmma_f32_16x16x32_f16(false, a1, false, b0, (short)0, acc[1][0], false, false);
        acc[1][1] = __builtin_amdgcn_wmma_f32_16x16x32_f16(false, a1, false, b1, (short)0, acc[1][1], false, false);
        acc[1][2] = __builtin_amdgcn_wmma_f32_16x16x32_f16(false, a1, false, b2, (short)0, acc[1][2], false, false);
        acc[1][3] = __builtin_amdgcn_wmma_f32_16x16x32_f16(false, a1, false, b3, (short)0, acc[1][3], false, false);

        // Next-tile copies must be complete (everywhere) before next iteration
        wait_async_zero();
        __syncthreads();
    }

    // D layout: vgpr r, lane l -> (M = r + 8*(l>=16), N = l&15)
    const int lane = threadIdx.x & 31;
    const int nn   = lane & 15;
    const int mb   = (lane >> 4) * 8;
#pragma unroll
    for (int i = 0; i < 2; ++i) {
#pragma unroll
        for (int j = 0; j < 4; ++j) {
#pragma unroll
            for (int r = 0; r < 8; ++r) {
                const int row = m0 + i * 16 + mb + r;
                const int col = n0 + j * 16 + nn;
                float v = acc[i][j][r] + bias[col];
                const size_t o = (size_t)row * Nd + col;
                if (EPI == 0) {
                    outH[o] = (_Float16)v;
                } else if (EPI == 1) {
                    outF[o] = v + residual[o];
                } else {                    // exact GELU: 0.5x(1+erf(x/sqrt(2)))
                    float g = 0.5f * v * (1.0f + erff(v * 0.70710678118654752f));
                    outH[o] = (_Float16)g;
                }
            }
        }
    }
}

// ---------------------------------------------------------------------------
// LayerNorm (f32 in -> f16 out), one block (256 thr) per token, C = 512
// ---------------------------------------------------------------------------
__global__ __launch_bounds__(256)
void layernorm_f16_kernel(const float* __restrict__ x,
                          const float* __restrict__ g,
                          const float* __restrict__ b,
                          _Float16*    __restrict__ out)
{
    const int row = blockIdx.x;
    const int tid = threadIdx.x;
    const float* xr = x + (size_t)row * C_;

    float vals[2];
    float s = 0.f, ss = 0.f;
#pragma unroll
    for (int i = 0; i < 2; ++i) {
        float v = xr[tid + i * 256];
        vals[i] = v; s += v; ss += v * v;
    }
#pragma unroll
    for (int o = 16; o > 0; o >>= 1) {
        s  += __shfl_xor(s,  o, 32);
        ss += __shfl_xor(ss, o, 32);
    }
    __shared__ float sb[8], ssb[8];
    const int lane = tid & 31, wid = tid >> 5;
    if (lane == 0) { sb[wid] = s; ssb[wid] = ss; }
    __syncthreads();
    if (tid == 0) {
        float ts = 0.f, tss = 0.f;
#pragma unroll
        for (int i = 0; i < 8; ++i) { ts += sb[i]; tss += ssb[i]; }
        sb[0] = ts; ssb[0] = tss;
    }
    __syncthreads();
    const float mu  = sb[0]  * (1.0f / C_);
    const float var = ssb[0] * (1.0f / C_) - mu * mu;
    const float inv = rsqrtf(var + 1e-5f);
#pragma unroll
    for (int i = 0; i < 2; ++i) {
        const int c = tid + i * 256;
        out[(size_t)row * C_ + c] = (_Float16)((vals[i] - mu) * inv * g[c] + b[c]);
    }
}

// ---------------------------------------------------------------------------
// 1D dilated neighborhood attention (NATTEN), thread per (n,h,t).
// qkv: f16 [NT, 3C]  (q|k|v per row, head channels h*64..h*64+63)
// ctx: f16 [NT, C]
// ---------------------------------------------------------------------------
__global__ __launch_bounds__(256)
void natten1d_kernel(const _Float16* __restrict__ qkv,
                     const float*    __restrict__ rpb,
                     _Float16*       __restrict__ ctx)
{
    const int idx = blockIdx.x * 256 + threadIdx.x;     // over N*H*T
    const int t = idx % T_;
    const int h = (idx / T_) % H_;
    const int n = idx / (T_ * H_);

    const int gdil = t % DIL_;
    const int li   = t / DIL_;
    int start = li - (KSZ_ / 2);
    if (start < 0) start = 0;
    if (start > L_ - KSZ_) start = L_ - KSZ_;

    const size_t rstride = 3 * C_;
    const _Float16* qp    = qkv + (size_t)(n * T_ + t) * rstride + h * DH_;
    const _Float16* kbase = qkv + C_ + h * DH_;
    const _Float16* vbase = qkv + 2 * C_ + h * DH_;

    size_t rows[KSZ_];
#pragma unroll
    for (int j = 0; j < KSZ_; ++j)
        rows[j] = (size_t)(n * T_ + gdil + (start + j) * DIL_) * rstride;

    float s[KSZ_];
#pragma unroll
    for (int j = 0; j < KSZ_; ++j) s[j] = 0.f;

    for (int c = 0; c < DH_; ++c) {
        const float q = (float)qp[c];
#pragma unroll
        for (int j = 0; j < KSZ_; ++j)
            s[j] += q * (float)kbase[rows[j] + c];
    }

    const float scale = 0.125f;                          // 1/sqrt(64)
    float mx = -3.0e38f;
#pragma unroll
    for (int j = 0; j < KSZ_; ++j) {
        s[j] = s[j] * scale + rpb[h * (2 * KSZ_ - 1) + (start + j - li + KSZ_ - 1)];
        mx = fmaxf(mx, s[j]);
    }
    float den = 0.f;
#pragma unroll
    for (int j = 0; j < KSZ_; ++j) { s[j] = __expf(s[j] - mx); den += s[j]; }
    const float rden = 1.f / den;
#pragma unroll
    for (int j = 0; j < KSZ_; ++j) s[j] *= rden;

    _Float16* op = ctx + (size_t)(n * T_ + t) * C_ + h * DH_;
    for (int c = 0; c < DH_; ++c) {
        float o = 0.f;
#pragma unroll
        for (int j = 0; j < KSZ_; ++j)
            o += s[j] * (float)vbase[rows[j] + c];
        op[c] = (_Float16)o;
    }
}

// ---------------------------------------------------------------------------
// Conversion helpers
// ---------------------------------------------------------------------------
__global__ __launch_bounds__(256)
void f32_to_f16_kernel(const float* __restrict__ src, _Float16* __restrict__ dst, int n) {
    const int i = blockIdx.x * 256 + threadIdx.x;
    if (i < n) dst[i] = (_Float16)src[i];
}

__global__ __launch_bounds__(256)
void concat3_f32_kernel(const float* __restrict__ a, const float* __restrict__ b,
                        const float* __restrict__ c, float* __restrict__ dst, int n) {
    const int i = blockIdx.x * 256 + threadIdx.x;
    if (i < n) { dst[i] = a[i]; dst[n + i] = b[i]; dst[2 * n + i] = c[i]; }
}

// ---------------------------------------------------------------------------
// Host-side launcher
// ---------------------------------------------------------------------------
extern "C" void kernel_launch(void* const* d_in, const int* in_sizes, int n_in,
                              void* d_out, int out_size, void* d_ws, size_t ws_size,
                              hipStream_t stream)
{
    const float* x     = (const float*)d_in[0];   // [N,T,C]
    const float* ln1_g = (const float*)d_in[1];
    const float* ln1_b = (const float*)d_in[2];
    const float* wq    = (const float*)d_in[3];
    const float* bq    = (const float*)d_in[4];
    const float* wk    = (const float*)d_in[5];
    const float* bk    = (const float*)d_in[6];
    const float* wv    = (const float*)d_in[7];
    const float* bv    = (const float*)d_in[8];
    const float* rpb   = (const float*)d_in[9];
    const float* wo    = (const float*)d_in[10];
    const float* bo    = (const float*)d_in[11];
    const float* ln2_g = (const float*)d_in[12];
    const float* ln2_b = (const float*)d_in[13];
    const float* w1    = (const float*)d_in[14];
    const float* b1    = (const float*)d_in[15];
    const float* w2    = (const float*)d_in[16];
    const float* b2    = (const float*)d_in[17];

    // Workspace layout (256B aligned)
    char* ws = (char*)d_ws;
    size_t off = 0;
    auto take = [&](size_t bytes) -> char* {
        char* p = ws + off;
        off = (off + bytes + 255) & ~(size_t)255;
        return p;
    };
    _Float16* h16    = (_Float16*)take((size_t)NT_ * C_ * 2);        // LN1 output
    _Float16* qkv16  = (_Float16*)take((size_t)NT_ * 3 * C_ * 2);    // q|k|v
    _Float16* ctx16  = (_Float16*)take((size_t)NT_ * C_ * 2);        // attention out
    float*    hidden = (float*)   take((size_t)NT_ * C_ * 4);        // post-Wo residual
    _Float16* y16    = (_Float16*)take((size_t)NT_ * C_ * 2);        // LN2 output
    _Float16* ff16   = (_Float16*)take((size_t)NT_ * FF_ * 2);       // GELU output
    _Float16* wqkv16 = (_Float16*)take((size_t)3 * C_ * C_ * 2);
    _Float16* wo16   = (_Float16*)take((size_t)C_ * C_ * 2);
    _Float16* w116   = (_Float16*)take((size_t)FF_ * C_ * 2);
    _Float16* w216   = (_Float16*)take((size_t)C_ * FF_ * 2);
    float*    bqkv   = (float*)   take((size_t)3 * C_ * 4);

    // ---- 0. weight / bias conversion -------------------------------------
    const int cc = C_ * C_;
    f32_to_f16_kernel<<<(cc + 255) / 256, 256, 0, stream>>>(wq, wqkv16,          cc);
    f32_to_f16_kernel<<<(cc + 255) / 256, 256, 0, stream>>>(wk, wqkv16 + cc,     cc);
    f32_to_f16_kernel<<<(cc + 255) / 256, 256, 0, stream>>>(wv, wqkv16 + 2 * cc, cc);
    f32_to_f16_kernel<<<(cc + 255) / 256, 256, 0, stream>>>(wo, wo16,            cc);
    f32_to_f16_kernel<<<(FF_ * C_ + 255) / 256, 256, 0, stream>>>(w1, w116, FF_ * C_);
    f32_to_f16_kernel<<<(C_ * FF_ + 255) / 256, 256, 0, stream>>>(w2, w216, C_ * FF_);
    concat3_f32_kernel<<<(C_ + 255) / 256, 256, 0, stream>>>(bq, bk, bv, bqkv, C_);

    // ---- 1. LN1 -----------------------------------------------------------
    layernorm_f16_kernel<<<NT_, 256, 0, stream>>>(x, ln1_g, ln1_b, h16);

    // ---- 2. fused QKV projection  [NT,512] x [1536,512]^T -----------------
    gemm_wmma_kernel<0><<<dim3(NT_ / BM_, (3 * C_) / BN_), 256, 0, stream>>>(
        h16, wqkv16, bqkv, nullptr, nullptr, qkv16, NT_, 3 * C_, C_);

    // ---- 3. dilated neighborhood attention --------------------------------
    natten1d_kernel<<<(N_ * H_ * T_) / 256, 256, 0, stream>>>(qkv16, rpb, ctx16);

    // ---- 4. Wo projection + residual -> hidden ----------------------------
    gemm_wmma_kernel<1><<<dim3(NT_ / BM_, C_ / BN_), 256, 0, stream>>>(
        ctx16, wo16, bo, x, hidden, nullptr, NT_, C_, C_);

    // ---- 5. LN2 -----------------------------------------------------------
    layernorm_f16_kernel<<<NT_, 256, 0, stream>>>(hidden, ln2_g, ln2_b, y16);

    // ---- 6. W1 + exact GELU ----------------------------------------------
    gemm_wmma_kernel<2><<<dim3(NT_ / BM_, FF_ / BN_), 256, 0, stream>>>(
        y16, w116, b1, nullptr, nullptr, ff16, NT_, FF_, C_);

    // ---- 7. W2 + bias + residual -> out -----------------------------------
    gemm_wmma_kernel<1><<<dim3(NT_ / BM_, C_ / BN_), 256, 0, stream>>>(
        ff16, w216, b2, hidden, (float*)d_out, nullptr, NT_, C_, FF_);

    (void)in_sizes; (void)n_in; (void)out_size; (void)ws_size;
}
